// LayerDepthValueReprojNormedDualQAttention_33621003993525
// MI455X (gfx1250) — matrix operand order repaired and checked by reading, not compiled, not run on hardware
//
#include <hip/hip_runtime.h>
#include <stdint.h>

// ---------------------------------------------------------------------------
// Problem constants (fixed by the reference: B=1, S=2048, d=1024, H=16, hd=64, L=8)
// ---------------------------------------------------------------------------
#define S_TOK   2048
#define DMODEL  1024
#define NHEAD   16
#define HDIM    64
#define NLAYER  8
#define SL      (S_TOK * NLAYER)   // 16384
#define D3      (3 * DMODEL)      // 3072

typedef __attribute__((ext_vector_type(16))) __bf16 v16bf;
typedef __attribute__((ext_vector_type(8)))  __bf16 v8bf;
typedef __attribute__((ext_vector_type(8)))  float  v8f;
typedef __attribute__((ext_vector_type(4)))  unsigned int u32x4;
typedef __attribute__((ext_vector_type(4)))  int  i32x4;
typedef __attribute__((ext_vector_type(8)))  int  i32x8;

#define USE_TDM __has_builtin(__builtin_amdgcn_tensor_load_to_lds)

// fp32 -> bf16 (round-to-nearest-even), bit result
__device__ __forceinline__ unsigned short f2bf(float f) {
  union { float f; unsigned int u; } v; v.f = f;
  unsigned int u = v.u;
  unsigned int r = u + 0x7FFFu + ((u >> 16) & 1u);
  return (unsigned short)(r >> 16);
}

__device__ __forceinline__ v16bf make_frag(v8bf lo, v8bf hi) {
  return __builtin_shufflevector(lo, hi, 0,1,2,3,4,5,6,7,8,9,10,11,12,13,14,15);
}
// A-fragment (16-bit A 16x32): lane holds 16 contiguous K at p[0..15]
__device__ __forceinline__ v16bf load_frag_a(const __bf16* p) {
  const v8bf* q = (const v8bf*)p;
  return make_frag(q[0], q[1]);
}
// B-fragment: lane holds 16 contiguous K values
__device__ __forceinline__ v16bf load_frag_c16(const __bf16* p) {
  const v8bf* q = (const v8bf*)p;
  return make_frag(q[0], q[1]);
}

// CDNA5 async global->LDS copy (16B per lane), tracked by ASYNCcnt.
__device__ __forceinline__ void async_copy_b128(uint32_t lds_addr, const void* gaddr) {
  asm volatile("global_load_async_to_lds_b128 %0, %1, off"
               :: "v"(lds_addr), "v"(gaddr) : "memory");
}
__device__ __forceinline__ void wait_async0() {
  asm volatile("s_wait_asynccnt 0" ::: "memory");
}
__device__ __forceinline__ uint32_t lds_addr_of(const void* p) {
  return (uint32_t)(uintptr_t)p;   // LDS aperture: addr[31:0] is the LDS offset
}

#if USE_TDM
// Tensor Data Mover: DMA a 2D tile (rows x 32 bf16 elements, row stride =
// strideElems) from global memory into LDS (packed rows, 64B each).
// D# layout per CDNA5 ISA §8.3/§8.4: group0 = {flags, lds_addr, global_addr,
// type=2}; group1 = {data_size=2B, tensor_dim0/1, tile_dim0/1/2, dim0 stride}.
__device__ __forceinline__ void tdm_load_tile_2d(uint32_t lds_addr, const void* gaddr,
                                                 uint32_t rows, uint32_t strideElems) {
  const uint64_t ga = (uint64_t)(uintptr_t)gaddr;
  u32x4 g0;
  g0[0] = 1u;                                   // count=1 (valid user descriptor)
  g0[1] = lds_addr;                             // lds_addr[31:0]
  g0[2] = (uint32_t)ga;                         // global_addr[31:0]
  g0[3] = (uint32_t)(ga >> 32) | (2u << 30);    // global_addr[56:32] | type=2
  i32x8 g1;
  g1[0] = 0x00010000;                           // workgroup_mask=0, data_size=1 (2B)
  g1[1] = (int)(32u << 16);                     // tensor_dim0 = 32 (elems/row)
  g1[2] = (int)(rows << 16);                    // tensor_dim1 = rows
  g1[3] = (int)(32u << 16);                     // tile_dim0 = 32
  g1[4] = (int)rows;                            // tile_dim1 = rows, tile_dim2 = 0
  g1[5] = (int)strideElems;                     // tensor_dim0_stride[31:0]
  g1[6] = 0;
  g1[7] = 0;
  const i32x4 z4 = {0, 0, 0, 0};
#if __clang_major__ >= 23
  const i32x8 z8 = {0, 0, 0, 0, 0, 0, 0, 0};
  __builtin_amdgcn_tensor_load_to_lds(g0, g1, z4, z4, z8, 0);
#else
  __builtin_amdgcn_tensor_load_to_lds(g0, g1, z4, z4, 0);
#endif
}
#endif

// ---------------------------------------------------------------------------
// Elementwise fp32 -> bf16
// ---------------------------------------------------------------------------
__global__ __launch_bounds__(256) void to_bf16_kernel(const float* __restrict__ in,
                                                      unsigned short* __restrict__ out,
                                                      int n) {
  int i = blockIdx.x * 256 + threadIdx.x;
  if (i < n) out[i] = f2bf(in[i]);
}

// ---------------------------------------------------------------------------
// Reprojection gather + LayerNorm: past_values [L][1][H][S][hd] -> ri_bf16 [S*L][d]
// ---------------------------------------------------------------------------
__global__ __launch_bounds__(256) void ln_reproj_kernel(const float* __restrict__ pv,
                                                        unsigned short* __restrict__ rib) {
  const int r = blockIdx.x;          // 0..SL-1
  const int s = r >> 3;              // / NLAYER
  const int l = r & (NLAYER - 1);
  const int tid = threadIdx.x;
  const int wave = tid >> 5, lane = tid & 31;

  float vals[4];
  float sum = 0.f, sq = 0.f;
#pragma unroll
  for (int i = 0; i < 4; ++i) {
    int j = tid + i * 256;           // 0..1023
    int h = j >> 6, e = j & 63;
    float v = pv[(((size_t)l * NHEAD + h) * S_TOK + s) * HDIM + e];
    vals[i] = v; sum += v; sq += v * v;
  }
#pragma unroll
  for (int m = 16; m >= 1; m >>= 1) {
    sum += __shfl_xor(sum, m);
    sq  += __shfl_xor(sq, m);
  }
  __shared__ float ssum[8], ssq[8];
  if (lane == 0) { ssum[wave] = sum; ssq[wave] = sq; }
  __syncthreads();
  if (wave == 0) {
    float a = (lane < 8) ? ssum[lane] : 0.f;
    float b = (lane < 8) ? ssq[lane]  : 0.f;
#pragma unroll
    for (int m = 4; m >= 1; m >>= 1) { a += __shfl_xor(a, m); b += __shfl_xor(b, m); }
    if (lane == 0) { ssum[0] = a; ssq[0] = b; }
  }
  __syncthreads();
  const float mean = ssum[0] * (1.f / 1024.f);
  const float var  = ssq[0] * (1.f / 1024.f) - mean * mean;
  const float rstd = rsqrtf(var + 1e-5f);
#pragma unroll
  for (int i = 0; i < 4; ++i) {
    int j = tid + i * 256;
    rib[(size_t)r * DMODEL + j] = f2bf((vals[i] - mean) * rstd);
  }
}

// ---------------------------------------------------------------------------
// GEMM: C[M][N] = A[M][K](bf16) * B[N][K](bf16)^T + bias[N]  (fp32 out)
// 128x128 block tile, K step 32, double-buffered LDS.
// Tile staging: TDM descriptor DMA (wave 0, TENSORcnt) when available,
// else per-lane async global->LDS copies (ASYNCcnt).
// 8 waves; each wave: 16 rows x 128 cols = 8 WMMA accumulators.
// ---------------------------------------------------------------------------
__global__ __launch_bounds__(256) void gemm_bf16_wmma(const __bf16* __restrict__ A,
                                                      const __bf16* __restrict__ B,
                                                      const float* __restrict__ bias,
                                                      float* __restrict__ C,
                                                      int M, int N, int K) {
  __shared__ __bf16 As[2][128 * 32];   // 8 KB per buffer
  __shared__ __bf16 Bs[2][128 * 32];   // 8 KB per buffer
  const int tid  = threadIdx.x;
  const int wave = tid >> 5;
  const int lane = tid & 31;
  const int half = lane >> 4;
  const int l16  = lane & 15;
  const int bm = blockIdx.x * 128;
  const int bn = blockIdx.y * 128;

  v8f acc[8];
#pragma unroll
  for (int t = 0; t < 8; ++t)
#pragma unroll
    for (int i = 0; i < 8; ++i) acc[t][i] = 0.f;

  const int nk = K >> 5;

#if USE_TDM
  auto stage = [&](int k0, int buf) {
    if (wave == 0) {
      tdm_load_tile_2d(lds_addr_of(&As[buf][0]), A + (size_t)bm * K + k0, 128, (uint32_t)K);
      tdm_load_tile_2d(lds_addr_of(&Bs[buf][0]), B + (size_t)bn * K + k0, 128, (uint32_t)K);
    }
  };
#else
  // Staging map: 128x32 bf16 tile = 8 KB; 256 threads x 2 async b128 each.
  const int sr = tid >> 1;           // 0..127 (row in tile)
  const int sc = (tid & 1) << 4;     // 0 or 16 (col in tile)
  auto stage = [&](int k0, int buf) {
    const __bf16* ga = A + (size_t)(bm + sr) * K + k0 + sc;
    uint32_t la = lds_addr_of(&As[buf][sr * 32 + sc]);
    async_copy_b128(la,      ga);
    async_copy_b128(la + 16, ga + 8);
    const __bf16* gb = B + (size_t)(bn + sr) * K + k0 + sc;
    uint32_t lb = lds_addr_of(&Bs[buf][sr * 32 + sc]);
    async_copy_b128(lb,      gb);
    async_copy_b128(lb + 16, gb + 8);
  };
#endif

  stage(0, 0);

  for (int i = 0; i < nk; ++i) {
#if USE_TDM
    if (wave == 0) __builtin_amdgcn_s_wait_tensorcnt(0);  // buf(i) DMA landed
#else
    wait_async0();          // this wave's copies into buf(i) landed
#endif
    __syncthreads();        // buffer published; everyone done with buf(i-1)
    if (i + 1 < nk) stage((i + 1) << 5, (i + 1) & 1);  // overwrites buf(i-1): safe

    const int buf = i & 1;
    v16bf af = load_frag_a(&As[buf][(wave * 16 + l16) * 32 + (half ? 16 : 0)]);
    v16bf bf8[8];
#pragma unroll
    for (int t = 0; t < 8; ++t)
      bf8[t] = load_frag_c16(&Bs[buf][(t * 16 + l16) * 32 + (half ? 16 : 0)]);
#pragma unroll
    for (int t = 0; t < 8; ++t)
      acc[t] = __builtin_amdgcn_wmma_f32_16x16x32_bf16(false, af, false, bf8[t],
                                                       (short)0, acc[t], false, false);
    __syncthreads();        // all waves done reading buf(i) before next overwrite
  }

#pragma unroll
  for (int t = 0; t < 8; ++t) {
    int col = bn + t * 16 + l16;
    float bv = bias ? bias[col] : 0.f;
    float* crow = C + (size_t)(bm + wave * 16 + half * 8) * N + col;
#pragma unroll
    for (int vi = 0; vi < 8; ++vi)
      crow[(size_t)vi * N] = acc[t][vi] + bv;
  }
}

// ---------------------------------------------------------------------------
// Split fp32 qkv (S x 3d) into per-head bf16: qh/kh [H][S][64], vT [H][64][S]
// ---------------------------------------------------------------------------
__global__ __launch_bounds__(256) void split_qkv_kernel(const float* __restrict__ qkv,
                                                        unsigned short* __restrict__ qh,
                                                        unsigned short* __restrict__ kh,
                                                        unsigned short* __restrict__ vT) {
  int idx = blockIdx.x * 256 + threadIdx.x;  // < S*d
  int s = idx >> 10;
  int j = idx & 1023;
  int h = j >> 6, e = j & 63;
  const float* row = qkv + (size_t)s * D3;
  qh[((size_t)h * S_TOK + s) * HDIM + e] = f2bf(row[j]);
  kh[((size_t)h * S_TOK + s) * HDIM + e] = f2bf(row[DMODEL + j]);
  vT[((size_t)h * HDIM + e) * S_TOK + s] = f2bf(row[2 * DMODEL + j]);
}

// ---------------------------------------------------------------------------
// mem_scores[h][s][l] = scale * dot(q_col[s, h*64:], rk[s*L+l, h*64:])  (fp32)
// ---------------------------------------------------------------------------
__global__ __launch_bounds__(256) void memscore_kernel(const float* __restrict__ qcol,
                                                       const float* __restrict__ rk,
                                                       float* __restrict__ ms) {
  int idx = blockIdx.x * 256 + threadIdx.x;  // < H*S*L
  int l = idx & (NLAYER - 1);
  int s = (idx >> 3) & (S_TOK - 1);
  int h = idx >> 14;
  const float* q = qcol + (size_t)s * DMODEL + h * HDIM;
  const float* k = rk + ((size_t)s * NLAYER + l) * DMODEL + h * HDIM;
  float a = 0.f;
#pragma unroll 8
  for (int e = 0; e < HDIM; ++e) a += q[e] * k[e];
  ms[((size_t)h * S_TOK + s) * NLAYER + l] = a * 0.125f;
}

// ---------------------------------------------------------------------------
// Causal flash attention + 8 memory keys per token.
// grid = (S/128, H); 8 waves/block; each wave owns 16 query rows.
// ---------------------------------------------------------------------------
__global__ __launch_bounds__(256) void attn_kernel(const __bf16* __restrict__ qh,
                                                   const __bf16* __restrict__ kh,
                                                   const __bf16* __restrict__ vT,
                                                   const float* __restrict__ memsc,
                                                   const float* __restrict__ rv,
                                                   unsigned short* __restrict__ ctxb) {
  __shared__ __bf16 Pt[8][16 * 32];   // per-wave P bounce tile (C-layout -> A-frag)
  const int tid  = threadIdx.x;
  const int wave = tid >> 5;
  const int lane = tid & 31;
  const int half = lane >> 4;
  const int l16  = lane & 15;
  const int h    = blockIdx.y;
  const int qbw  = blockIdx.x * 128 + wave * 16;   // first query row of this wave
  const float scale = 0.125f;                      // 1/sqrt(64)

  // Q fragments for this wave's 16 rows (K = 0..31 and 32..63)
  const __bf16* qbase = qh + ((size_t)h * S_TOK + qbw) * HDIM;
  v16bf qf0 = load_frag_a(qbase + (size_t)l16 * HDIM + (half ? 16 : 0));
  v16bf qf1 = load_frag_a(qbase + (size_t)l16 * HDIM + 32 + (half ? 16 : 0));

  v8f acc[4];
  float m_i[8], l_i[8];
#pragma unroll
  for (int t = 0; t < 4; ++t)
#pragma unroll
    for (int i = 0; i < 8; ++i) acc[t][i] = 0.f;
#pragma unroll
  for (int i = 0; i < 8; ++i) { m_i[i] = -3.0e38f; l_i[i] = 0.f; }

  for (int kb = 0; kb < qbw + 16; kb += 32) {
    // ---- preload all K fragments (2 col tiles x 2 K-chunks), then grouped WMMAs
    v16bf kf0[2], kf1[2];
#pragma unroll
    for (int c = 0; c < 2; ++c) {
      const __bf16* kp = kh + ((size_t)h * S_TOK + kb + c * 16 + l16) * HDIM + (half ? 16 : 0);
      kf0[c] = load_frag_c16(kp);
      kf1[c] = load_frag_c16(kp + 32);
    }
    v8f st[2];
#pragma unroll
    for (int c = 0; c < 2; ++c) {
      v8f z;
#pragma unroll
      for (int i = 0; i < 8; ++i) z[i] = 0.f;
      z = __builtin_amdgcn_wmma_f32_16x16x32_bf16(false, qf0, false, kf0[c], (short)0, z, false, false);
      z = __builtin_amdgcn_wmma_f32_16x16x32_bf16(false, qf1, false, kf1[c], (short)0, z, false, false);
      st[c] = z;
    }

    // hoist V fragments so their global latency overlaps the softmax math
    v16bf vf[4];
#pragma unroll
    for (int t = 0; t < 4; ++t)
      vf[t] = load_frag_c16(vT + ((size_t)h * HDIM + t * 16 + l16) * S_TOK + kb + (half ? 16 : 0));

    const bool needMask = (kb + 31 > qbw);
#pragma unroll
    for (int c = 0; c < 2; ++c)
#pragma unroll
      for (int vi = 0; vi < 8; ++vi) {
        float sc = st[c][vi] * scale;
        if (needMask) {
          int keyg = kb + c * 16 + l16;
          int rowg = qbw + vi + half * 8;
          if (keyg > rowg) sc = -3.0e38f;
        }
        st[c][vi] = sc;
      }

    // ---- online softmax (each row lives across the 16 lanes of a half-wave)
    float alpha[8];
#pragma unroll
    for (int vi = 0; vi < 8; ++vi) {
      float cm = fmaxf(st[0][vi], st[1][vi]);
      cm = fmaxf(cm, __shfl_xor(cm, 1));
      cm = fmaxf(cm, __shfl_xor(cm, 2));
      cm = fmaxf(cm, __shfl_xor(cm, 4));
      cm = fmaxf(cm, __shfl_xor(cm, 8));
      float mn = fmaxf(m_i[vi], cm);
      alpha[vi] = __expf(m_i[vi] - mn);
      float p0 = __expf(st[0][vi] - mn);
      float p1 = __expf(st[1][vi] - mn);
      st[0][vi] = p0; st[1][vi] = p1;
      float rs = p0 + p1;
      rs += __shfl_xor(rs, 1);
      rs += __shfl_xor(rs, 2);
      rs += __shfl_xor(rs, 4);
      rs += __shfl_xor(rs, 8);
      l_i[vi] = l_i[vi] * alpha[vi] + rs;
      m_i[vi] = mn;
    }
#pragma unroll
    for (int t = 0; t < 4; ++t)
#pragma unroll
      for (int vi = 0; vi < 8; ++vi) acc[t][vi] *= alpha[vi];

    // ---- P: C-layout -> bf16 LDS -> A-fragment layout (wave-private, in-order DS)
    __bf16* pw = &Pt[wave][0];
#pragma unroll
    for (int vi = 0; vi < 8; ++vi) {
      int m = vi + half * 8;
      ((unsigned short*)pw)[m * 32 + l16]      = f2bf(st[0][vi]);
      ((unsigned short*)pw)[m * 32 + 16 + l16] = f2bf(st[1][vi]);
    }
    asm volatile("s_wait_dscnt 0" ::: "memory");
    v16bf pf = load_frag_a(pw + (size_t)l16 * 32 + (half ? 16 : 0));

    // ---- ctx += P(16x32) @ V(32x64), grouped WMMAs
#pragma unroll
    for (int t = 0; t < 4; ++t)
      acc[t] = __builtin_amdgcn_wmma_f32_16x16x32_bf16(false, pf, false, vf[t],
                                                       (short)0, acc[t], false, false);
  }

  // ---- merge the 8 per-token memory keys (fp32) and write normalized ctx (bf16)
#pragma unroll
  for (int vi = 0; vi < 8; ++vi) {
    const int m = vi + half * 8;
    const int s = qbw + m;
    const float* ms = memsc + ((size_t)h * S_TOK + s) * NLAYER;
    float mx = m_i[vi];
    float w[8];
#pragma unroll
    for (int l = 0; l < 8; ++l) { w[l] = ms[l]; mx = fmaxf(mx, w[l]); }
    float alpha = __expf(m_i[vi] - mx);
    float rs = 0.f;
#pragma unroll
    for (int l = 0; l < 8; ++l) { w[l] = __expf(w[l] - mx); rs += w[l]; }
    const float ln  = l_i[vi] * alpha + rs;
    const float inv = 1.f / ln;
    const float* rvbase = rv + ((size_t)s * NLAYER) * DMODEL + (size_t)h * HDIM;
#pragma unroll
    for (int t = 0; t < 4; ++t) {
      int col = t * 16 + l16;
      float a = acc[t][vi] * alpha;
#pragma unroll
      for (int l = 0; l < 8; ++l) a += w[l] * rvbase[(size_t)l * DMODEL + col];
      ctxb[(size_t)s * DMODEL + h * HDIM + col] = f2bf(a * inv);
    }
  }
}

// ---------------------------------------------------------------------------
// Host launcher
// ---------------------------------------------------------------------------
extern "C" void kernel_launch(void* const* d_in, const int* in_sizes, int n_in,
                              void* d_out, int out_size, void* d_ws, size_t ws_size,
                              hipStream_t stream) {
  const float* x    = (const float*)d_in[0];
  const float* pv   = (const float*)d_in[1];
  const float* Wqkv = (const float*)d_in[2];
  const float* bqkv = (const float*)d_in[3];
  const float* Wcol = (const float*)d_in[4];
  const float* bcol = (const float*)d_in[5];
  const float* Wout = (const float*)d_in[6];
  const float* bout = (const float*)d_in[7];
  float* out = (float*)d_out;

  char* ws = (char*)d_ws;
  size_t off = 0;
  auto alloc = [&](size_t bytes) -> char* {
    char* p = ws + off;
    off += (bytes + 255) & ~(size_t)255;
    return p;
  };

  __bf16* xb    = (__bf16*)alloc((size_t)S_TOK * DMODEL * 2);
  __bf16* Wqkvb = (__bf16*)alloc((size_t)D3 * DMODEL * 2);
  __bf16* Wcolb = (__bf16*)alloc((size_t)DMODEL * DMODEL * 2);
  __bf16* Woutb = (__bf16*)alloc((size_t)DMODEL * DMODEL * 2);
  __bf16* rib   = (__bf16*)alloc((size_t)SL * DMODEL * 2);
  float*  qkv   = (float*) alloc((size_t)S_TOK * D3 * 4);
  float*  qcol  = (float*) alloc((size_t)S_TOK * DMODEL * 4);
  float*  rk    = (float*) alloc((size_t)SL * DMODEL * 4);
  float*  rvb   = (float*) alloc((size_t)SL * DMODEL * 4);
  __bf16* qhb   = (__bf16*)alloc((size_t)NHEAD * S_TOK * HDIM * 2);
  __bf16* khb   = (__bf16*)alloc((size_t)NHEAD * S_TOK * HDIM * 2);
  __bf16* vTb   = (__bf16*)alloc((size_t)NHEAD * HDIM * S_TOK * 2);
  float*  ms    = (float*) alloc((size_t)NHEAD * S_TOK * NLAYER * 4);
  __bf16* ctxb  = (__bf16*)alloc((size_t)S_TOK * DMODEL * 2);

  // 1) fp32 -> bf16 conversions
  to_bf16_kernel<<<(S_TOK * DMODEL) / 256, 256, 0, stream>>>(x, (unsigned short*)xb, S_TOK * DMODEL);
  to_bf16_kernel<<<(D3 * DMODEL) / 256, 256, 0, stream>>>(Wqkv, (unsigned short*)Wqkvb, D3 * DMODEL);
  to_bf16_kernel<<<(DMODEL * DMODEL) / 256, 256, 0, stream>>>(Wcol, (unsigned short*)Wcolb, DMODEL * DMODEL);
  to_bf16_kernel<<<(DMODEL * DMODEL) / 256, 256, 0, stream>>>(Wout, (unsigned short*)Woutb, DMODEL * DMODEL);

  // 2) reprojection gather + LayerNorm
  ln_reproj_kernel<<<SL, 256, 0, stream>>>(pv, (unsigned short*)rib);

  // 3) WMMA GEMMs (TDM / async-LDS double-buffered)
  gemm_bf16_wmma<<<dim3(S_TOK / 128, D3 / 128), 256, 0, stream>>>(xb, Wqkvb, bqkv, qkv, S_TOK, D3, DMODEL);
  gemm_bf16_wmma<<<dim3(S_TOK / 128, DMODEL / 128), 256, 0, stream>>>(xb, Wcolb, bcol, qcol, S_TOK, DMODEL, DMODEL);
  gemm_bf16_wmma<<<dim3(SL / 128, DMODEL / 128), 256, 0, stream>>>(
      rib, Wqkvb + (size_t)DMODEL * DMODEL, bqkv + DMODEL, rk, SL, DMODEL, DMODEL);
  gemm_bf16_wmma<<<dim3(SL / 128, DMODEL / 128), 256, 0, stream>>>(
      rib, Wqkvb + (size_t)2 * DMODEL * DMODEL, bqkv + 2 * DMODEL, rvb, SL, DMODEL, DMODEL);

  // 4) head split / V transpose, memory scores
  split_qkv_kernel<<<(S_TOK * DMODEL) / 256, 256, 0, stream>>>(
      qkv, (unsigned short*)qhb, (unsigned short*)khb, (unsigned short*)vTb);
  memscore_kernel<<<(NHEAD * S_TOK * NLAYER) / 256, 256, 0, stream>>>(qcol, rk, ms);

  // 5) attention (causal tokens + 8 memory keys)
  attn_kernel<<<dim3(S_TOK / 128, NHEAD), 256, 0, stream>>>(qhb, khb, vTb, ms, rvb, (unsigned short*)ctxb);

  // 6) output projection
  gemm_bf16_wmma<<<dim3(S_TOK / 128, DMODEL / 128), 256, 0, stream>>>(ctxb, Woutb, bout, out, S_TOK, DMODEL, DMODEL);
}
